// Model_37589553774958
// MI455X (gfx1250) — matrix operand order repaired and checked by reading, not compiled, not run on hardware
//
#include <hip/hip_runtime.h>

// ---------------------------------------------------------------------------
// Problem constants (match reference)
// ---------------------------------------------------------------------------
#define BSZ   256
#define HDIM  1024
#define VSZ   50000
#define NLAY  2
#define THREE_H (3 * HDIM)

// ---------------------------------------------------------------------------
// Types
// ---------------------------------------------------------------------------
typedef __attribute__((ext_vector_type(16))) __bf16          bf16x16;
typedef __attribute__((ext_vector_type(8)))  float           f32x8;
typedef __attribute__((ext_vector_type(4)))  float           f32x4;
typedef unsigned short                                        u16;
typedef unsigned int                                          u32;
typedef __attribute__((ext_vector_type(8)))  unsigned short  u16x8;
typedef __attribute__((ext_vector_type(16))) unsigned short  u16x16;
typedef __attribute__((ext_vector_type(8)))  unsigned int    u32x8;

__device__ __forceinline__ u16 f2bf_bits(float f) {
  u32 u = __builtin_bit_cast(u32, f);
  return (u16)(u >> 16);                      // truncate fp32 -> bf16
}
__device__ __forceinline__ float sigmoidf_fast(float x) {
  return 1.0f / (1.0f + __expf(-x));
}
__device__ __forceinline__ float tanhf_fast(float x) {
  float e = __expf(2.0f * x);
  return (e - 1.0f) / (e + 1.0f);
}

// Pack {hi16(b) : hi16(a)} into one dword with a single v_perm_b32.
// V_PERM sel: bytes 0-3 select from S1, 4-7 from S0. dst = {b[3],b[2],a[3],a[2]}.
__device__ __forceinline__ u32 pack2_bf16(float a, float b) {
  return __builtin_amdgcn_perm(__builtin_bit_cast(u32, b),
                               __builtin_bit_cast(u32, a), 0x07060302u);
}

// A fragment (16x32 bf16, row-major source with leading dim HDIM).
// Lane m (0..15): K = [0..15]; lane m+16: K = [16..31].
// Caller passes p = base + row*HDIM + k0 + (lane>=16 ? 16 : 0).
// Two b128 loads + register renaming only (no VALU shuffles).
__device__ __forceinline__ bf16x16 load_a_bf16(const u16* __restrict__ p) {
  u16x8 lo = *(const u16x8*)(p);
  u16x8 hi = *(const u16x8*)(p + 8);
  u16x16 c = __builtin_shufflevector(lo, hi, 0, 1, 2, 3, 4, 5, 6, 7,
                                             8, 9, 10, 11, 12, 13, 14, 15);
  return __builtin_bit_cast(bf16x16, c);
}

// B fragment (32x16 bf16) from fp32 row-major weights [N][K]: column n of B is
// row n of W (contiguous in k). Lane n (0..15): K=[0..15]; lane n+16: K=[16..31].
// Caller passes p = W + n*K + k0 + (lane>=16 ? 16 : 0).
// 4 b128 loads + 8 v_perm_b32.
__device__ __forceinline__ bf16x16 load_b_f32(const float* __restrict__ p) {
  f32x4 v0 = ((const f32x4*)p)[0];
  f32x4 v1 = ((const f32x4*)p)[1];
  f32x4 v2 = ((const f32x4*)p)[2];
  f32x4 v3 = ((const f32x4*)p)[3];
  u32x8 r;
  r[0] = pack2_bf16(v0[0], v0[1]);
  r[1] = pack2_bf16(v0[2], v0[3]);
  r[2] = pack2_bf16(v1[0], v1[1]);
  r[3] = pack2_bf16(v1[2], v1[3]);
  r[4] = pack2_bf16(v2[0], v2[1]);
  r[5] = pack2_bf16(v2[2], v2[3]);
  r[6] = pack2_bf16(v3[0], v3[1]);
  r[7] = pack2_bf16(v3[2], v3[3]);
  return __builtin_bit_cast(bf16x16, r);
}

__device__ __forceinline__ f32x8 splat8(float v) {
  f32x8 r;
#pragma unroll
  for (int i = 0; i < 8; ++i) r[i] = v;
  return r;
}

// ---------------------------------------------------------------------------
// Kernel 1: embedding gather + fp32 -> bf16 conversion of x and hidden[0..1]
// ---------------------------------------------------------------------------
__global__ __launch_bounds__(256)
void prep_kernel(const int* __restrict__ ids,
                 const float* __restrict__ emb,       // [V][H]
                 const float* __restrict__ hidden,    // [L][B][H]
                 u16* __restrict__ x_bf16,            // [B][H]
                 u16* __restrict__ h_bf16) {          // [L][B][H]
  int i = blockIdx.x * blockDim.x + threadIdx.x;
  if (i >= BSZ * HDIM) return;
  int b = i >> 10;          // / HDIM
  int h = i & (HDIM - 1);   // % HDIM
  x_bf16[i] = f2bf_bits(emb[(size_t)ids[b] * HDIM + h]);
  h_bf16[i] = f2bf_bits(hidden[i]);
  h_bf16[BSZ * HDIM + i] = f2bf_bits(hidden[BSZ * HDIM + i]);
}

// ---------------------------------------------------------------------------
// Kernel 2: fused GRU layer.
// Block tile: 256(B) x 16(n). 8 waves; wave w owns rows [32w, 32w+32).
// Per gate: gi = x @ w_ih[g]^T + b_ih, gh = h @ w_hh[g]^T + b_hh via WMMA.
// ---------------------------------------------------------------------------
__global__ __launch_bounds__(256)
void gru_layer_kernel(const u16*  __restrict__ xA,      // [B][H] bf16 layer input
                      const u16*  __restrict__ hA,      // [B][H] bf16 hidden
                      const float* __restrict__ h_old,  // [B][H] fp32 hidden
                      const float* __restrict__ w_ih,   // [3H][H]
                      const float* __restrict__ w_hh,   // [3H][H]
                      const float* __restrict__ b_ih,   // [3H]
                      const float* __restrict__ b_hh,   // [3H]
                      float* __restrict__ h_new_f32,    // [B][H] -> d_out slice
                      u16*   __restrict__ h_new_bf16) { // [B][H] next layer A
  const int n0    = blockIdx.x * 16;          // 64 blocks cover H
  const int wave  = threadIdx.x >> 5;
  const int lane  = threadIdx.x & 31;
  const int half  = lane >> 4;
  const int l15   = lane & 15;
  const int mbase = wave * 32;

  const int aoff = half * 16;   // A-fragment K sub-offset
  const int boff = half * 16;   // B-fragment K sub-offset

  f32x8 rr[2], zz[2], nn[2];

#pragma unroll
  for (int g = 0; g < 3; ++g) {
    const float bi = b_ih[g * HDIM + n0 + l15];
    const float bh = b_hh[g * HDIM + n0 + l15];
    f32x8 ai[2], ah[2];
#pragma unroll
    for (int ms = 0; ms < 2; ++ms) { ai[ms] = splat8(bi); ah[ms] = splat8(bh); }

    const float* wih_row = w_ih + (size_t)(g * HDIM + n0 + l15) * HDIM;
    const float* whh_row = w_hh + (size_t)(g * HDIM + n0 + l15) * HDIM;

    for (int k0 = 0; k0 < HDIM; k0 += 32) {
      bf16x16 bih = load_b_f32(wih_row + k0 + boff);
      bf16x16 bhh = load_b_f32(whh_row + k0 + boff);
#pragma unroll
      for (int ms = 0; ms < 2; ++ms) {
        const int m = mbase + ms * 16 + l15;
        bf16x16 ax  = load_a_bf16(xA + (size_t)m * HDIM + k0 + aoff);
        bf16x16 ahh = load_a_bf16(hA + (size_t)m * HDIM + k0 + aoff);
        ai[ms] = __builtin_amdgcn_wmma_f32_16x16x32_bf16(
            false, ax,  false, bih, (short)0, ai[ms], false, false);
        ah[ms] = __builtin_amdgcn_wmma_f32_16x16x32_bf16(
            false, ahh, false, bhh, (short)0, ah[ms], false, false);
      }
    }

    if (g == 0) {
#pragma unroll
      for (int ms = 0; ms < 2; ++ms)
#pragma unroll
        for (int j = 0; j < 8; ++j)
          rr[ms][j] = sigmoidf_fast(ai[ms][j] + ah[ms][j]);
    } else if (g == 1) {
#pragma unroll
      for (int ms = 0; ms < 2; ++ms)
#pragma unroll
        for (int j = 0; j < 8; ++j)
          zz[ms][j] = sigmoidf_fast(ai[ms][j] + ah[ms][j]);
    } else {
#pragma unroll
      for (int ms = 0; ms < 2; ++ms)
#pragma unroll
        for (int j = 0; j < 8; ++j)
          nn[ms][j] = tanhf_fast(ai[ms][j] + rr[ms][j] * ah[ms][j]);
    }
  }

  // h' = (1-z)*n + z*h   (C/D layout: component j -> row mtile + 8*half + j,
  //                        col n0 + l15)
  const int col = n0 + l15;
#pragma unroll
  for (int ms = 0; ms < 2; ++ms) {
    const int rbase = mbase + ms * 16 + half * 8;
#pragma unroll
    for (int j = 0; j < 8; ++j) {
      const int row = rbase + j;
      const float ho = h_old[(size_t)row * HDIM + col];
      const float hv = (1.0f - zz[ms][j]) * nn[ms][j] + zz[ms][j] * ho;
      h_new_f32[(size_t)row * HDIM + col]  = hv;
      h_new_bf16[(size_t)row * HDIM + col] = f2bf_bits(hv);
    }
  }
}

// ---------------------------------------------------------------------------
// Kernel 3: decoder GEMM  logits[256,V] = x[256,H] @ dec_w[V,H]^T + dec_b.
// Block tile 256(M) x 64(N): full M per block => each dec_w element read once
// from HBM (8x re-read served by WGP$/L2).
// Wave: 2 M-subtiles x 4 N-subtiles = 8 accumulators, 256 WMMAs per wave.
// ---------------------------------------------------------------------------
__global__ __launch_bounds__(256)
void decoder_kernel(const u16*  __restrict__ xA,      // [B][H] bf16
                    const float* __restrict__ dec_w,  // [V][H]
                    const float* __restrict__ dec_b,  // [V]
                    float* __restrict__ out) {        // [B][V]
  const int n0    = blockIdx.x * 64;
  const int wave  = threadIdx.x >> 5;
  const int lane  = threadIdx.x & 31;
  const int half  = lane >> 4;
  const int l15   = lane & 15;
  const int mbase = wave * 32;
  const int aoff  = half * 16;
  const int boff  = half * 16;

  // Clamped column indices keep EXEC uniform; stores are guarded at the end.
  int vcs[4];
#pragma unroll
  for (int ns = 0; ns < 4; ++ns) {
    int v = n0 + ns * 16 + l15;
    vcs[ns] = v < VSZ ? v : VSZ - 1;
  }

  f32x8 acc[2][4];
#pragma unroll
  for (int ns = 0; ns < 4; ++ns) {
    float bias = dec_b[vcs[ns]];
#pragma unroll
    for (int ms = 0; ms < 2; ++ms) acc[ms][ns] = splat8(bias);
  }

  for (int k0 = 0; k0 < HDIM; k0 += 32) {
    bf16x16 a[2];
#pragma unroll
    for (int ms = 0; ms < 2; ++ms)
      a[ms] = load_a_bf16(xA + (size_t)(mbase + ms * 16 + l15) * HDIM + k0 + aoff);
#pragma unroll
    for (int ns = 0; ns < 4; ++ns) {
      const float* bp = dec_w + (size_t)vcs[ns] * HDIM + k0 + boff;
      // Prefetch the weight stream two K-steps ahead (global_prefetch path).
      if (k0 + 64 < HDIM) __builtin_prefetch(bp + 64, 0, 1);
      bf16x16 bfr = load_b_f32(bp);
#pragma unroll
      for (int ms = 0; ms < 2; ++ms)
        acc[ms][ns] = __builtin_amdgcn_wmma_f32_16x16x32_bf16(
            false, a[ms], false, bfr, (short)0, acc[ms][ns], false, false);
    }
  }

#pragma unroll
  for (int ns = 0; ns < 4; ++ns) {
    const int v = n0 + ns * 16 + l15;
    if (v < VSZ) {
#pragma unroll
      for (int ms = 0; ms < 2; ++ms) {
        const int rbase = mbase + ms * 16 + half * 8;
#pragma unroll
        for (int j = 0; j < 8; ++j)
          out[(size_t)(rbase + j) * VSZ + v] = acc[ms][ns][j];
      }
    }
  }
}

// ---------------------------------------------------------------------------
// Launch
// ---------------------------------------------------------------------------
extern "C" void kernel_launch(void* const* d_in, const int* in_sizes, int n_in,
                              void* d_out, int out_size, void* d_ws, size_t ws_size,
                              hipStream_t stream) {
  const int*   ids    = (const int*)  d_in[0];
  const float* hidden = (const float*)d_in[1];
  const float* emb    = (const float*)d_in[2];
  const float* w_ih   = (const float*)d_in[3];
  const float* w_hh   = (const float*)d_in[4];
  const float* b_ih   = (const float*)d_in[5];
  const float* b_hh   = (const float*)d_in[6];
  const float* dec_w  = (const float*)d_in[7];
  const float* dec_b  = (const float*)d_in[8];

  float* logits     = (float*)d_out;                       // [B][V]
  float* hidden_out = logits + (size_t)BSZ * VSZ;          // [L][B][H]

  // Workspace: xbuf0 | xbuf1 | hbuf[L]  (all bf16, stored as u16)
  u16* xbuf0 = (u16*)d_ws;
  u16* xbuf1 = xbuf0 + (size_t)BSZ * HDIM;
  u16* hbuf  = xbuf1 + (size_t)BSZ * HDIM;

  prep_kernel<<<(BSZ * HDIM + 255) / 256, 256, 0, stream>>>(
      ids, emb, hidden, xbuf0, hbuf);

  // Layer 0: x = emb gather, h = hidden[0]; h_new -> hidden_out[0], bf16 -> xbuf1
  gru_layer_kernel<<<HDIM / 16, 256, 0, stream>>>(
      xbuf0, hbuf, hidden,
      w_ih, w_hh, b_ih, b_hh,
      hidden_out, xbuf1);

  // Layer 1: x = layer0 out, h = hidden[1]; h_new -> hidden_out[1], bf16 -> xbuf0
  gru_layer_kernel<<<HDIM / 16, 256, 0, stream>>>(
      xbuf1, hbuf + (size_t)BSZ * HDIM, hidden + (size_t)BSZ * HDIM,
      w_ih + (size_t)THREE_H * HDIM, w_hh + (size_t)THREE_H * HDIM,
      b_ih + THREE_H, b_hh + THREE_H,
      hidden_out + (size_t)BSZ * HDIM, xbuf0);

  // Decoder: logits = x @ dec_w^T + dec_b
  decoder_kernel<<<(VSZ + 63) / 64, 256, 0, stream>>>(
      xbuf0, dec_w, dec_b, logits);
}
